// BoxTransformerModule_5093831213381
// MI455X (gfx1250) — matrix operand order repaired and checked by reading, not compiled, not run on hardware
//
#include <hip/hip_runtime.h>
#include <hip/hip_bf16.h>

// ---------------------------------------------------------------------------
// BoxTransformer forward on gfx1250 (MI455X): bf16 WMMA everywhere, fp32 accum.
// B=8, S=512, N=513 (CLS prepended), padded to NP=544. D=256, H=8, HD=32,
// C=2, P=64, L=4.
// ---------------------------------------------------------------------------

#define BB   8
#define SS   512
#define NN   513         // S+1
#define NP   544         // padded rows (17 * 32)
#define DD   256
#define HH   8
#define HD   32
#define PP   64
#define LL   4
#define RROWS (BB*NP)    // 4352
#define ATT_SCALE 0.17677669529663687f   // 1/sqrt(32)

typedef __attribute__((ext_vector_type(16))) __bf16 bf16x16;
typedef __attribute__((ext_vector_type(8)))  float  f32x8;

union Frag16 { bf16x16 v; uint4 q[2]; unsigned short u[16]; };

static __device__ __forceinline__ unsigned short f2bf(float f) {
    unsigned int u = __builtin_bit_cast(unsigned int, f);
    unsigned int r = (u + 0x7FFFu + ((u >> 16) & 1u)) >> 16;   // RNE
    return (unsigned short)r;
}
static __device__ __forceinline__ float bf2f(unsigned short s) {
    return __builtin_bit_cast(float, (unsigned int)s << 16);
}

// ---------------------------------------------------------------------------
// Build x = [empty_embed ; data ; zeros], fp32 + bf16, plus bf16 pos_embed.
// ---------------------------------------------------------------------------
__global__ void k_build(const float* __restrict__ data, const float* __restrict__ empty,
                        const float* __restrict__ pos, float* __restrict__ x,
                        unsigned short* __restrict__ xb, unsigned short* __restrict__ peb) {
    int r = blockIdx.x, t = threadIdx.x;
    if (r < RROWS) {
        int b = r / NP, i = r % NP;
        float v = 0.f;
        if (i == 0) v = empty[t];
        else if (i <= SS) v = data[((size_t)b * SS + (i - 1)) * DD + t];
        x[(size_t)r * DD + t]  = v;
        xb[(size_t)r * DD + t] = f2bf(v);
    } else {
        int p = r - RROWS;
        peb[(size_t)p * DD + t] = f2bf(pos[(size_t)p * DD + t]);
    }
}

// ---------------------------------------------------------------------------
// One-time weight prep: WT[n][k] (bf16) = W[k][n] (f32). Tiled transpose,
// coalesced on both sides. grid = (K/32, N/32), 256 threads.
// ---------------------------------------------------------------------------
__global__ void __launch_bounds__(256) k_wt(const float* __restrict__ W,
                                            unsigned short* __restrict__ WT, int K, int N) {
    __shared__ float tile[32][33];
    int k0 = blockIdx.x * 32, n0 = blockIdx.y * 32;
    int tr = threadIdx.x >> 5, tc = threadIdx.x & 31;
    for (int r = tr; r < 32; r += 8)
        tile[r][tc] = W[(size_t)(k0 + r) * N + n0 + tc];
    __syncthreads();
    for (int r = tr; r < 32; r += 8)
        WT[(size_t)(n0 + r) * K + k0 + tc] = f2bf(tile[tc][r]);
}

// ---------------------------------------------------------------------------
// GEMM: out = act(A_bf16[MxK] @ WT_bf16[N][K]^T + bias).  No LDS, no barriers:
// A and B fragments are direct contiguous b128 global loads (weights L2-hot).
// Block = 256 thr = 8 waves (4 row-groups x 2 col-groups), 32x32 per wave,
// 128x64 per block, 4 WMMAs per k-step.
// ---------------------------------------------------------------------------
__global__ void __launch_bounds__(256) k_gemm(const unsigned short* __restrict__ A,
                                              const unsigned short* __restrict__ WT,
                                              const float* __restrict__ bias,
                                              float* __restrict__ outF,
                                              unsigned short* __restrict__ outB,
                                              int M, int K, int N, int act) {
    const int tid = threadIdx.x;
    const int lane = tid & 31, wave = tid >> 5;
    const int wr = wave & 3, wc = wave >> 2;
    const int g = lane >> 4, ln = lane & 15;
    const int m0 = blockIdx.x * 128 + wr * 32;
    const int n0 = blockIdx.y * 64 + wc * 32;

    int r0 = m0 + ln;      if (r0 > M - 1) r0 = M - 1;   // row clamp for short M
    int r1 = m0 + 16 + ln; if (r1 > M - 1) r1 = M - 1;
    const unsigned short* a0p = A + (size_t)r0 * K;
    const unsigned short* a1p = A + (size_t)r1 * K;
    const unsigned short* b0p = WT + (size_t)(n0 + ln) * K;
    const unsigned short* b1p = WT + (size_t)(n0 + 16 + ln) * K;

    f32x8 acc00 = {0,0,0,0,0,0,0,0}, acc01 = {0,0,0,0,0,0,0,0};
    f32x8 acc10 = {0,0,0,0,0,0,0,0}, acc11 = {0,0,0,0,0,0,0,0};

    #pragma unroll 4
    for (int kk = 0; kk < K; kk += 32) {
        Frag16 a0, a1, b0, b1;
        a0.q[0] = *(const uint4*)(a0p + kk + 8 * g);
        a0.q[1] = *(const uint4*)(a0p + kk + 16 + 8 * g);
        a1.q[0] = *(const uint4*)(a1p + kk + 8 * g);
        a1.q[1] = *(const uint4*)(a1p + kk + 16 + 8 * g);
        b0.q[0] = *(const uint4*)(b0p + kk + 8 * g);
        b0.q[1] = *(const uint4*)(b0p + kk + 16 + 8 * g);
        b1.q[0] = *(const uint4*)(b1p + kk + 8 * g);
        b1.q[1] = *(const uint4*)(b1p + kk + 16 + 8 * g);
        acc00 = __builtin_amdgcn_wmma_f32_16x16x32_bf16(false, a0.v, false, b0.v, (short)0, acc00, false, false);
        acc01 = __builtin_amdgcn_wmma_f32_16x16x32_bf16(false, a0.v, false, b1.v, (short)0, acc01, false, false);
        acc10 = __builtin_amdgcn_wmma_f32_16x16x32_bf16(false, a1.v, false, b0.v, (short)0, acc10, false, false);
        acc11 = __builtin_amdgcn_wmma_f32_16x16x32_bf16(false, a1.v, false, b1.v, (short)0, acc11, false, false);
    }

    for (int ai = 0; ai < 2; ++ai) {
        for (int bi = 0; bi < 2; ++bi) {
            f32x8 acc = ai ? (bi ? acc11 : acc10) : (bi ? acc01 : acc00);
            int n = n0 + bi * 16 + ln;
            float bs = bias ? bias[n] : 0.f;
            for (int r = 0; r < 8; ++r) {
                int m = m0 + ai * 16 + r + 8 * g;
                if (m >= M) continue;
                float v = acc[r] + bs;
                if (act == 1) v = 0.5f * v * (1.f + erff(v * 0.70710678118654752f)); // exact gelu
                size_t o = (size_t)m * N + n;
                if (outF) outF[o] = v;
                if (outB) outB[o] = f2bf(v);
            }
        }
    }
}

// ---------------------------------------------------------------------------
// Positional score tables: c2p[b,i,c,p,h] = sum_d qc*pk ; p2c[b,j,c,p,h].
// ---------------------------------------------------------------------------
__global__ void k_pos_tables(const unsigned short* __restrict__ qb,
                             const unsigned short* __restrict__ kb,
                             const float* __restrict__ pkf, const float* __restrict__ pqf,
                             float* __restrict__ c2p, float* __restrict__ p2c) {
    __shared__ float qr[DD], kr[DD];
    int row = blockIdx.x, t = threadIdx.x;
    qr[t] = bf2f(qb[(size_t)row * DD + t]);
    kr[t] = bf2f(kb[(size_t)row * DD + t]);
    __syncthreads();
    for (int e = t; e < 1024; e += 256) {                // C*P*H = 1024
        int c = e >> 9, p = (e >> 3) & 63, hh = e & 7;
        int base = hh * HD + c * 16;
        const float* pk = pkf + (size_t)p * DD + base;
        const float* pq = pqf + (size_t)p * DD + base;
        float s1 = 0.f, s2 = 0.f;
        for (int d = 0; d < 16; ++d) { s1 += qr[base + d] * pk[d]; s2 += pq[d] * kr[base + d]; }
        size_t o = (size_t)row * 1024 + (size_t)c * 512 + (size_t)p * 8 + hh;
        c2p[o] = s1; p2c[o] = s2;
    }
}

// ---------------------------------------------------------------------------
// Fused attention: block = 4 waves = 4 heads, 16-query tile.
// Pass 1: scores (WMMA) + pos gather + mask -> LDS (16 x 544 f32).
// Softmax: all 32 lanes (row = lane&15, half = lane>>4), float2 LDS loads,
//   shfl_xor(16) cross-half combine, packed 2xbf16 ds_store_b32 of probs.
// Pass 2: P@V via WMMA; A-frag = 2 ds_load_b128 of bf16 probs; V transposed in LDS.
// rel_pos (int2) + mask tiles staged once per block, shared by all 4 heads.
// ---------------------------------------------------------------------------
#define SC_F   (16*548)          // score rows, f32
#define VT_F   512               // v-tile transposed, 1024 bf16
#define RS_F   16                // row sums
#define PB_F   (16*552/2)        // prob rows, bf16 (stride 552)
#define WAVE_F (SC_F + VT_F + RS_F + PB_F)   // 13712 floats per wave

__global__ void __launch_bounds__(128) k_attn(const unsigned short* __restrict__ qb,
                                              const unsigned short* __restrict__ kb,
                                              const unsigned short* __restrict__ vb,
                                              const int* __restrict__ rel,
                                              const unsigned char* __restrict__ mask,
                                              const float* __restrict__ c2p,
                                              const float* __restrict__ p2c,
                                              float* __restrict__ upd) {
    extern __shared__ char smem[];
    const int tid = threadIdx.x, lane = tid & 31, wave = tid >> 5;
    const int b = blockIdx.y;
    const int i0 = blockIdx.x * 16;
    const int h = blockIdx.z * 4 + wave;
    const int g = lane >> 4, ln = lane & 15;

    float* wb = (float*)smem + wave * WAVE_F;
    float* sc = wb;                                             // [16][548] f32
    unsigned short* vt = (unsigned short*)(wb + SC_F);          // [32][32] bf16
    float* rowsum = wb + SC_F + VT_F;                           // [16]
    unsigned short* pbuf = (unsigned short*)(wb + SC_F + VT_F + RS_F); // [16][552] bf16
    int2* srp = (int2*)((float*)smem + 4 * WAVE_F);             // [16][32] rel pairs
    int* smask = (int*)(srp + 512);                             // [32] key-ok flags

    // Q fragment (constant across the key loop): rows i0..i0+15, K = HD = 32
    Frag16 aq;
    {
        const unsigned short* qp = qb + ((size_t)b * NP + i0 + ln) * DD + h * HD;
        aq.q[0] = *(const uint4*)(qp + 8 * g);
        aq.q[1] = *(const uint4*)(qp + 16 + 8 * g);
    }

    const int2* rel2 = (const int2*)rel;    // both coordinates of one (i,j) pair

    // ---- pass 1: scores ----
    for (int j0 = 0; j0 < NP; j0 += 32) {
        for (int e = tid; e < 512; e += 128) {              // shared rel tile (int2)
            int im = e >> 5, jn = e & 31;
            int gi = i0 + im, gj = j0 + jn;
            int2 v = {0, 0};
            if (gi >= 1 && gi < NN && gj >= 1 && gj < NN)
                v = rel2[((size_t)b * SS + (gi - 1)) * SS + (gj - 1)];
            srp[e] = v;
        }
        if (tid < 32) {                                     // shared key-valid tile
            int j = j0 + tid;
            smask[tid] = (j < NN) && (j == 0 || mask[(size_t)b * SS + (j - 1)] != 0);
        }
        __syncthreads();
        for (int jh = 0; jh < 32; jh += 16) {
            Frag16 bk;
            const unsigned short* kp = kb + ((size_t)b * NP + j0 + jh + ln) * DD + h * HD + 8 * g;
            bk.q[0] = *(const uint4*)(kp); bk.q[1] = *(const uint4*)(kp + 16);
            f32x8 s8 = {0,0,0,0,0,0,0,0};
            s8 = __builtin_amdgcn_wmma_f32_16x16x32_bf16(false, aq.v, false, bk.v, (short)0, s8, false, false);
            for (int r = 0; r < 8; ++r) {
                int m = r + 8 * g;
                int i = i0 + m;
                int jn = jh + ln;
                int j = j0 + jn;
                float s = s8[r];
                if (i >= 1 && i < NN && j >= 1 && j < NN) {
                    int2 rp = srp[m * 32 + jn];
                    size_t bi = ((size_t)b * NP + i) * 1024;
                    size_t bj = ((size_t)b * NP + j) * 1024;
                    s += c2p[bi + (size_t)rp.x * 8 + h] + p2c[bj + (size_t)rp.x * 8 + h]
                       + c2p[bi + 512 + (size_t)rp.y * 8 + h] + p2c[bj + 512 + (size_t)rp.y * 8 + h];
                }
                s *= ATT_SCALE;
                sc[m * 548 + j0 + jn] = smask[jn] ? s : -1e9f;
            }
        }
        __syncthreads();
    }

    // ---- softmax: all lanes active, row = ln, contiguous half = g ----
    {
        float* row = sc + ln * 548;
        const int jb = g * 272;
        float mx = -1e30f;
        for (int t = 0; t < 136; ++t) {
            float2 v = *(const float2*)(row + jb + 2 * t);
            mx = fmaxf(mx, fmaxf(v.x, v.y));
        }
        mx = fmaxf(mx, __shfl_xor(mx, 16, 32));             // combine halves
        float sum = 0.f;
        unsigned int* pb32 = (unsigned int*)(pbuf + ln * 552) + g * 136;
        for (int t = 0; t < 136; ++t) {
            float2 v = *(const float2*)(row + jb + 2 * t);
            float e0 = __expf(v.x - mx), e1 = __expf(v.y - mx);
            sum += e0 + e1;
            pb32[t] = (unsigned int)f2bf(e0) | ((unsigned int)f2bf(e1) << 16);
        }
        sum += __shfl_xor(sum, 16, 32);
        if (lane < 16) rowsum[ln] = sum;
    }
    __syncthreads();

    // ---- pass 2: P @ V ----
    f32x8 o0 = {0,0,0,0,0,0,0,0};
    f32x8 o1 = {0,0,0,0,0,0,0,0};
    const unsigned short* pbrow = pbuf + ln * 552;
    for (int j0 = 0; j0 < NP; j0 += 32) {
        {   // stage V tile transposed: vt[d][j]
            const unsigned short* vp = vb + ((size_t)b * NP + j0 + lane) * DD + h * HD;
            union { uint4 q[4]; unsigned short u[32]; } vr;
            vr.q[0] = *(const uint4*)(vp);
            vr.q[1] = *(const uint4*)(vp + 8);
            vr.q[2] = *(const uint4*)(vp + 16);
            vr.q[3] = *(const uint4*)(vp + 24);
            for (int d = 0; d < 32; ++d) vt[d * 32 + lane] = vr.u[d];
        }
        __syncthreads();
        Frag16 aw;                                          // bf16 probs, 2 x b128
        aw.q[0] = *(const uint4*)(pbrow + j0 + 16 * g);
        aw.q[1] = *(const uint4*)(pbrow + j0 + 16 * g + 8);
        {
            Frag16 bv;
            const unsigned short* vtp = &vt[ln * 32 + 16 * g];
            bv.q[0] = *(const uint4*)(vtp); bv.q[1] = *(const uint4*)(vtp + 8);
            o0 = __builtin_amdgcn_wmma_f32_16x16x32_bf16(false, aw.v, false, bv.v, (short)0, o0, false, false);
        }
        {
            Frag16 bv;
            const unsigned short* vtp = &vt[(16 + ln) * 32 + 16 * g];
            bv.q[0] = *(const uint4*)(vtp); bv.q[1] = *(const uint4*)(vtp + 8);
            o1 = __builtin_amdgcn_wmma_f32_16x16x32_bf16(false, aw.v, false, bv.v, (short)0, o1, false, false);
        }
        __syncthreads();
    }
    for (int r = 0; r < 8; ++r) {
        int m = r + 8 * g;
        int i = i0 + m;
        float inv = 1.f / rowsum[m];
        size_t o = ((size_t)b * NP + i) * DD + h * HD;
        upd[o + ln]      = o0[r] * inv;
        upd[o + 16 + ln] = o1[r] * inv;
    }
}

// ---------------------------------------------------------------------------
// x = LayerNorm(x + upd*res) ; writes fp32 + bf16. Padded rows forced to 0 so
// no garbage/NaN can leak into later WMMA tiles through K/V columns.
// ---------------------------------------------------------------------------
__global__ void k_res_ln(const float* __restrict__ x, const float* __restrict__ u,
                         const float* __restrict__ res, const float* __restrict__ gamma,
                         const float* __restrict__ beta, float* __restrict__ xo,
                         unsigned short* __restrict__ xbo) {
    __shared__ float red[256];
    int row = blockIdx.x, t = threadIdx.x;
    int i = row % NP;
    size_t o = (size_t)row * DD + t;
    if (i >= NN) { xo[o] = 0.f; xbo[o] = 0; return; }      // uniform per block
    float v = x[o] + u[o] * res[0];
    red[t] = v; __syncthreads();
    for (int s = 128; s > 0; s >>= 1) { if (t < s) red[t] += red[t + s]; __syncthreads(); }
    float mu = red[0] * (1.f / DD); __syncthreads();
    float d = v - mu;
    red[t] = d * d; __syncthreads();
    for (int s = 128; s > 0; s >>= 1) { if (t < s) red[t] += red[t + s]; __syncthreads(); }
    float var = red[0] * (1.f / DD);
    float out = d * rsqrtf(var + 1e-5f) * gamma[t] + beta[t];
    xo[o] = out; xbo[o] = f2bf(out);
}

// drop CLS + padding: out[b,s,:] = x[b, 1+s, :]
__global__ void k_out(const float* __restrict__ x, float* __restrict__ out) {
    int r = blockIdx.x, t = threadIdx.x;
    int b = r >> 9, s = r & 511;
    out[(size_t)r * DD + t] = x[((size_t)b * NP + 1 + s) * DD + t];
}

// ---------------------------------------------------------------------------
extern "C" void kernel_launch(void* const* d_in, const int* in_sizes, int n_in,
                              void* d_out, int out_size, void* d_ws, size_t ws_size,
                              hipStream_t stream) {
    const float*         data  = (const float*)d_in[0];
    const unsigned char* mask  = (const unsigned char*)d_in[1];
    const int*           rel   = (const int*)d_in[2];
    const float*         empty = (const float*)d_in[3];
    const float*         pos   = (const float*)d_in[4];
    const float *Wq = (const float*)d_in[5],  *bq  = (const float*)d_in[6];
    const float *Wk = (const float*)d_in[7],  *bk  = (const float*)d_in[8];
    const float *Wv = (const float*)d_in[9],  *bv  = (const float*)d_in[10];
    const float *Wpq= (const float*)d_in[11], *bpq = (const float*)d_in[12];
    const float *Wpk= (const float*)d_in[13], *bpk = (const float*)d_in[14];
    const float *res1= (const float*)d_in[15];
    const float *ln1g= (const float*)d_in[16], *ln1b= (const float*)d_in[17];
    const float *W1 = (const float*)d_in[18], *b1  = (const float*)d_in[19];
    const float *W2 = (const float*)d_in[20], *b2  = (const float*)d_in[21];
    const float *res2= (const float*)d_in[22];
    const float *ln2g= (const float*)d_in[23], *ln2b= (const float*)d_in[24];

    char* ws = (char*)d_ws;
    size_t off = 0;
    auto alloc = [&](size_t bytes) { char* p = ws + off; off = (off + bytes + 255) & ~(size_t)255; return p; };
    float*          x    = (float*)alloc((size_t)RROWS * DD * 4);
    unsigned short* xb   = (unsigned short*)alloc((size_t)RROWS * DD * 2);
    unsigned short* qbuf = (unsigned short*)alloc((size_t)RROWS * DD * 2);
    unsigned short* kbuf = (unsigned short*)alloc((size_t)RROWS * DD * 2);
    unsigned short* vbuf = (unsigned short*)alloc((size_t)RROWS * DD * 2);
    float*          upd  = (float*)alloc((size_t)RROWS * DD * 4);
    unsigned short* h1b  = (unsigned short*)alloc((size_t)RROWS * 512 * 2);
    float*          ffo  = (float*)alloc((size_t)RROWS * DD * 4);
    unsigned short* peb  = (unsigned short*)alloc((size_t)PP * DD * 2);
    float*          pkf  = (float*)alloc((size_t)PP * DD * 4);
    float*          pqf  = (float*)alloc((size_t)PP * DD * 4);
    float*          c2p  = (float*)alloc((size_t)RROWS * 1024 * 4);
    float*          p2c  = (float*)alloc((size_t)RROWS * 1024 * 4);
    // pre-transposed bf16 weights: [L][N][K]
    unsigned short* wqT  = (unsigned short*)alloc((size_t)LL * 256 * DD * 2);
    unsigned short* wkT  = (unsigned short*)alloc((size_t)LL * 256 * DD * 2);
    unsigned short* wvT  = (unsigned short*)alloc((size_t)LL * 256 * DD * 2);
    unsigned short* wpqT = (unsigned short*)alloc((size_t)LL * 256 * DD * 2);
    unsigned short* wpkT = (unsigned short*)alloc((size_t)LL * 256 * DD * 2);
    unsigned short* w1T  = (unsigned short*)alloc((size_t)LL * 512 * DD * 2);
    unsigned short* w2T  = (unsigned short*)alloc((size_t)LL * DD * 512 * 2);

    const int attnShmem = (4 * WAVE_F + 1024 + 32 + 8) * 4;   // ~224 KB dynamic LDS
    (void)hipFuncSetAttribute(reinterpret_cast<const void*>(k_attn),
                              hipFuncAttributeMaxDynamicSharedMemorySize, attnShmem);

    // one-time weight transpose+convert (L2-hot afterwards)
    for (int l = 0; l < LL; ++l) {
        const size_t wO = (size_t)l * DD * 256;
        k_wt<<<dim3(8, 8),  256, 0, stream>>>(Wq  + wO, wqT  + wO, DD, 256);
        k_wt<<<dim3(8, 8),  256, 0, stream>>>(Wk  + wO, wkT  + wO, DD, 256);
        k_wt<<<dim3(8, 8),  256, 0, stream>>>(Wv  + wO, wvT  + wO, DD, 256);
        k_wt<<<dim3(8, 8),  256, 0, stream>>>(Wpq + wO, wpqT + wO, DD, 256);
        k_wt<<<dim3(8, 8),  256, 0, stream>>>(Wpk + wO, wpkT + wO, DD, 256);
        k_wt<<<dim3(8, 16), 256, 0, stream>>>(W1 + (size_t)l * DD * 512, w1T + (size_t)l * DD * 512, DD, 512);
        k_wt<<<dim3(16, 8), 256, 0, stream>>>(W2 + (size_t)l * 512 * DD, w2T + (size_t)l * 512 * DD, 512, 256);
    }

    k_build<<<RROWS + PP, 256, 0, stream>>>(data, empty, pos, x, xb, peb);

    for (int l = 0; l < LL; ++l) {
        const size_t wO = (size_t)l * DD * 256, bO = (size_t)l * 256;
        // QKV projections (bf16 out only)
        k_gemm<<<dim3(34, 4), 256, 0, stream>>>(xb, wqT + wO, bq + bO, nullptr, qbuf, RROWS, DD, 256, 0);
        k_gemm<<<dim3(34, 4), 256, 0, stream>>>(xb, wkT + wO, bk + bO, nullptr, kbuf, RROWS, DD, 256, 0);
        k_gemm<<<dim3(34, 4), 256, 0, stream>>>(xb, wvT + wO, bv + bO, nullptr, vbuf, RROWS, DD, 256, 0);
        // positional projections (fp32 out)
        k_gemm<<<dim3(1, 4), 256, 0, stream>>>(peb, wpkT + wO, bpk + bO, pkf, nullptr, 64, DD, 256, 0);
        k_gemm<<<dim3(1, 4), 256, 0, stream>>>(peb, wpqT + wO, bpq + bO, pqf, nullptr, 64, DD, 256, 0);
        // bucket score tables
        k_pos_tables<<<RROWS, 256, 0, stream>>>(qbuf, kbuf, pkf, pqf, c2p, p2c);
        // fused attention: (i-tiles=33) x B x (2 head-groups of 4)
        k_attn<<<dim3(33, BB, 2), 128, attnShmem, stream>>>(qbuf, kbuf, vbuf, rel, mask, c2p, p2c, upd);
        // residual + LN 1
        k_res_ln<<<RROWS, 256, 0, stream>>>(x, upd, res1 + l, ln1g + bO, ln1b + bO, x, xb);
        // FFN
        k_gemm<<<dim3(34, 8), 256, 0, stream>>>(xb, w1T + (size_t)l * DD * 512, b1 + (size_t)l * 512,
                                                nullptr, h1b, RROWS, DD, 512, 1);
        k_gemm<<<dim3(34, 4), 256, 0, stream>>>(h1b, w2T + (size_t)l * 512 * DD, b2 + bO,
                                                ffo, nullptr, RROWS, 512, 256, 0);
        // residual + LN 2
        k_res_ln<<<RROWS, 256, 0, stream>>>(x, ffo, res2 + l, ln2g + bO, ln2b + bO, x, xb);
    }
    k_out<<<BB * SS, 256, 0, stream>>>(x, (float*)d_out);
}
